// RestormerTeacher_59287728554582
// MI455X (gfx1250) — compile-verified, matrix operand order checked by reading, not run on hardware
//
#include <hip/hip_runtime.h>
#include <hip/hip_bf16.h>
#include <math.h>

// ---------------------------------------------------------------------------
// Types for CDNA5 WMMA (wave32): v_wmma_f32_16x16x32_bf16
// ---------------------------------------------------------------------------
typedef __attribute__((ext_vector_type(16))) __bf16 v16bf;
typedef __attribute__((ext_vector_type(8)))  __bf16 v8bf;   // one 16B chunk
typedef __attribute__((ext_vector_type(8)))  float  v8f;

__device__ __forceinline__ __bf16 f2bf(float f) { return (__bf16)f; }

// Per-lane WMMA operand fragment: two contiguous 16-byte loads matching the
// ISA 16-bit operand striping (lanes 0-15: K {0..7,16..23}; 16-31: {8..15,24..31}).
__device__ __forceinline__ v16bf load_frag(const __bf16* p)
{
    v8bf lo = *(const v8bf*)(p);
    v8bf hi = *(const v8bf*)(p + 16);
    return __builtin_shufflevector(lo, hi, 0, 1, 2, 3, 4, 5, 6, 7,
                                   8, 9, 10, 11, 12, 13, 14, 15);
}

// ---------------------------------------------------------------------------
// Batched GEMM:  C[b](MxN) = A[b](Mp x Kp, bf16, row-major)
//                          * Bt[b](N x Kp, bf16, row-major)^T
// Operands pre-padded (A rows to 16, K to 32) and zero-filled: NO bounds
// checks in the hot loop. Each wave computes a 32x32 C tile (2x2 sub-tiles,
// 4 accumulators): 4 WMMAs per 8x16B loads -> 2x arithmetic intensity vs
// 16x16 tiling, and 4 independent accumulators (no WMMA->WMMA RAW chain).
// Edge sub-blocks: pointers clamped to block 0 (uniform, branch-free);
// duplicate results dropped by store guards.
// ---------------------------------------------------------------------------
__global__ __launch_bounds__(256)
void k_gemm_bf16(const __bf16* __restrict__ A, const __bf16* __restrict__ Bt,
                 float* __restrict__ C, int M, int N, int Kp,
                 long sAb, long sBb, long sCb)
{
    const int wave = threadIdx.x >> 5;
    const int lane = threadIdx.x & 31;
    const int nT32 = (N + 31) >> 5;
    const int tn = blockIdx.x * 8 + wave;
    const int tm = blockIdx.y;
    if (tn >= nT32) return;

    const __bf16* Ab = A  + (long)blockIdx.z * sAb;
    const __bf16* Bb = Bt + (long)blockIdx.z * sBb;
    float*        Cb = C  + (long)blockIdx.z * sCb;

    const int row = lane & 15;
    const int hi  = lane >> 4;
    const int Mp  = (M + 15) & ~15;

    const int m0 = tm * 32, m1 = m0 + 16;
    const int n0 = tn * 32, n1 = n0 + 16;
    const int m1c = (m1 < Mp) ? m1 : m0;   // clamp: duplicate work, discarded at store
    const int n1c = (n1 < N)  ? n1 : n0;

    const __bf16* a0p = Ab + (long)(m0  + row) * Kp + hi * 8;
    const __bf16* a1p = Ab + (long)(m1c + row) * Kp + hi * 8;
    const __bf16* b0p = Bb + (long)(n0  + row) * Kp + hi * 8;
    const __bf16* b1p = Bb + (long)(n1c + row) * Kp + hi * 8;

    v8f acc00 = {}, acc01 = {}, acc10 = {}, acc11 = {};
    for (int k0 = 0; k0 < Kp; k0 += 32) {
        v16bf a0 = load_frag(a0p + k0);
        v16bf a1 = load_frag(a1p + k0);
        v16bf b0 = load_frag(b0p + k0);
        v16bf b1 = load_frag(b1p + k0);
        __builtin_prefetch(a0p + k0 + 64, 0, 1);   // global_prefetch_b8
        __builtin_prefetch(a1p + k0 + 64, 0, 1);
        __builtin_prefetch(b0p + k0 + 64, 0, 1);
        __builtin_prefetch(b1p + k0 + 64, 0, 1);
        acc00 = __builtin_amdgcn_wmma_f32_16x16x32_bf16(false, a0, false, b0, (short)0, acc00, false, false);
        acc01 = __builtin_amdgcn_wmma_f32_16x16x32_bf16(false, a0, false, b1, (short)0, acc01, false, false);
        acc10 = __builtin_amdgcn_wmma_f32_16x16x32_bf16(false, a1, false, b0, (short)0, acc10, false, false);
        acc11 = __builtin_amdgcn_wmma_f32_16x16x32_bf16(false, a1, false, b1, (short)0, acc11, false, false);
    }
#pragma unroll
    for (int r = 0; r < 8; ++r) {
        const int mA = m0 + r + hi * 8;    // C/D layout: VGPR r -> M=r / 8+r
        const int mB = m1 + r + hi * 8;
        const int nA = n0 + row;
        const int nB = n1 + row;
        if (mA < M && nA < N) Cb[(long)mA * N + nA] = acc00[r];
        if (mA < M && nB < N) Cb[(long)mA * N + nB] = acc01[r];
        if (mB < M && nA < N) Cb[(long)mB * N + nA] = acc10[r];
        if (mB < M && nB < N) Cb[(long)mB * N + nB] = acc11[r];
    }
}

// ---------------------------------------------------------------------------
// fp32 (R x K) -> bf16 (Rp x Kp) zero-padded, same orientation (coalesced).
// ---------------------------------------------------------------------------
__global__ void k_pad_convert_bf16(const float* __restrict__ in, __bf16* __restrict__ out,
                                   int R, int K, int Rp, int Kp)
{
    long i = (long)blockIdx.x * blockDim.x + threadIdx.x;
    long total = (long)Rp * Kp;
    if (i >= total) return;
    int k = (int)(i % Kp);
    int r = (int)(i / Kp);
    out[i] = (r < R && k < K) ? f2bf(in[(long)r * K + k]) : (__bf16)0.0f;
}

// ---------------------------------------------------------------------------
// fp32 (C x N) -> bf16 (N x Kp), LDS-tiled 16x16 transpose (coalesced both
// sides, [16][17] padding avoids LDS bank conflicts). N multiple of 16.
// grid: (N/16, Kp/16), block: (16,16)
// ---------------------------------------------------------------------------
__global__ __launch_bounds__(256)
void k_transpose_convert_bf16(const float* __restrict__ in, __bf16* __restrict__ out,
                              int C, long N, int Kp)
{
    __shared__ float tile[16][17];
    const int  tx = threadIdx.x, ty = threadIdx.y;
    const long n0 = (long)blockIdx.x * 16;
    const int  c0 = blockIdx.y * 16;
    const int  c  = c0 + ty;
    tile[ty][tx] = (c < C) ? in[(long)c * N + (n0 + tx)] : 0.0f;
    __syncthreads();
    out[(n0 + ty) * Kp + (c0 + tx)] = f2bf(tile[tx][ty]);
}

// ---------------------------------------------------------------------------
// per-head transpose: v (heads*hd x N) fp32 -> out (heads, N, Kp) bf16, K pad.
// grid: (N/16, Kp/16, heads), block: (16,16)
// ---------------------------------------------------------------------------
__global__ __launch_bounds__(256)
void k_transpose_heads_bf16(const float* __restrict__ v, __bf16* __restrict__ out,
                            int heads, int hd, long N, int Kp)
{
    __shared__ float tile[16][17];
    const int  tx = threadIdx.x, ty = threadIdx.y;
    const long n0 = (long)blockIdx.x * 16;
    const int  c0 = blockIdx.y * 16;
    const int  h  = blockIdx.z;
    const int  d  = c0 + ty;
    tile[ty][tx] = (d < hd) ? v[((long)h * hd + d) * N + (n0 + tx)] : 0.0f;
    __syncthreads();
    out[((long)h * N + (n0 + ty)) * Kp + (c0 + tx)] = f2bf(tile[tx][ty]);
}

// ---------------------------------------------------------------------------
// Depthwise 3x3 conv, pad=1, stride=1. Weight (C,1,3,3).
// ---------------------------------------------------------------------------
__global__ void k_dwconv3x3(const float* __restrict__ in, const float* __restrict__ w,
                            float* __restrict__ out, int C, int H, int W)
{
    long i = (long)blockIdx.x * blockDim.x + threadIdx.x;
    long total = (long)C * H * W;
    if (i >= total) return;
    int x = (int)(i % W);
    int y = (int)((i / W) % H);
    int c = (int)(i / ((long)W * H));
    const float* wc = w + (long)c * 9;
    const float* ic = in + (long)c * H * W;
    float s = 0.0f;
#pragma unroll
    for (int ky = 0; ky < 3; ++ky) {
        int iy = y + ky - 1;
        if (iy < 0 || iy >= H) continue;
#pragma unroll
        for (int kx = 0; kx < 3; ++kx) {
            int ix = x + kx - 1;
            if (ix < 0 || ix >= W) continue;
            s += ic[(long)iy * W + ix] * wc[ky * 3 + kx];
        }
    }
    out[i] = s;
}

// ---------------------------------------------------------------------------
// Generic direct conv (patch 3x3, down 4x4 s2, upsample 3x3, out 3x3).
// ---------------------------------------------------------------------------
__global__ void k_conv2d(const float* __restrict__ in, const float* __restrict__ w,
                         float* __restrict__ out, int Ci, int Co, int Hi, int Wi,
                         int Kh, int Kw, int stride, int pad, int Ho, int Wo)
{
    long i = (long)blockIdx.x * blockDim.x + threadIdx.x;
    long total = (long)Co * Ho * Wo;
    if (i >= total) return;
    int ox = (int)(i % Wo);
    int oy = (int)((i / Wo) % Ho);
    int co = (int)(i / ((long)Wo * Ho));
    float s = 0.0f;
    for (int ci = 0; ci < Ci; ++ci) {
        const float* wp = w + (((long)co * Ci + ci) * Kh) * Kw;
        const float* ip = in + (long)ci * Hi * Wi;
        for (int ky = 0; ky < Kh; ++ky) {
            int iy = oy * stride - pad + ky;
            if (iy < 0 || iy >= Hi) continue;
            for (int kx = 0; kx < Kw; ++kx) {
                int ix = ox * stride - pad + kx;
                if (ix < 0 || ix >= Wi) continue;
                s += ip[(long)iy * Wi + ix] * wp[ky * Kw + kx];
            }
        }
    }
    out[i] = s;
}

// ---------------------------------------------------------------------------
// Bias-free channel LayerNorm: out = w[c] * x / sqrt(var_c + 1e-5)
// ---------------------------------------------------------------------------
__global__ void k_ln2d(const float* __restrict__ x, const float* __restrict__ wgt,
                       float* __restrict__ out, int C, long HW)
{
    long n = (long)blockIdx.x * blockDim.x + threadIdx.x;
    if (n >= HW) return;
    float sum = 0.0f, sq = 0.0f;
    for (int c = 0; c < C; ++c) {
        float v = x[(long)c * HW + n];
        sum += v; sq += v * v;
    }
    float mean = sum / C;
    float var  = sq / C - mean * mean;
    float inv  = rsqrtf(var + 1e-5f);
    for (int c = 0; c < C; ++c)
        out[(long)c * HW + n] = wgt[c] * x[(long)c * HW + n] * inv;
}

// ---------------------------------------------------------------------------
// Row-wise L2 normalization over N; writes bf16 directly (GEMM operand).
// ---------------------------------------------------------------------------
__global__ __launch_bounds__(256)
void k_l2norm_row_bf16(const float* __restrict__ in, __bf16* __restrict__ out, long Nlen)
{
    long row = blockIdx.x;
    const float* r = in + row * Nlen;
    __bf16* o = out + row * Nlen;
    __shared__ float red[256];
    float s = 0.0f;
    for (long i = threadIdx.x; i < Nlen; i += 256) { float v = r[i]; s += v * v; }
    red[threadIdx.x] = s;
    __syncthreads();
    for (int off = 128; off > 0; off >>= 1) {
        if ((int)threadIdx.x < off) red[threadIdx.x] += red[threadIdx.x + off];
        __syncthreads();
    }
    float inv = 1.0f / fmaxf(sqrtf(red[0]), 1e-12f);
    for (long i = threadIdx.x; i < Nlen; i += 256) o[i] = f2bf(r[i] * inv);
}

// ---------------------------------------------------------------------------
// softmax(attn * (softplus(log_temp[h]) + 1e-6)) over last axis; rows len D.
// ---------------------------------------------------------------------------
__global__ void k_softmax_temp(float* __restrict__ attn, const float* __restrict__ log_temp,
                               int heads, int D)
{
    int row = blockIdx.x * blockDim.x + threadIdx.x;
    if (row >= heads * D) return;
    int h = row / D;
    float temp = log1pf(expf(log_temp[h])) + 1e-6f;
    float* r = attn + (long)row * D;
    float mx = -3.4e38f;
    for (int j = 0; j < D; ++j) mx = fmaxf(mx, r[j] * temp);
    float s = 0.0f;
    for (int j = 0; j < D; ++j) { float e = expf(r[j] * temp - mx); r[j] = e; s += e; }
    float inv = 1.0f / s;
    for (int j = 0; j < D; ++j) r[j] *= inv;
}

// ---------------------------------------------------------------------------
// GDFN gate: out = gelu_exact(x1) * x2 ; y holds [x1(hid), x2(hid)].
// ---------------------------------------------------------------------------
__global__ void k_gelu_mul(const float* __restrict__ y, float* __restrict__ out,
                           int hid, long HW)
{
    long i = (long)blockIdx.x * blockDim.x + threadIdx.x;
    long total = (long)hid * HW;
    if (i >= total) return;
    float a = y[i];
    float b = y[i + total];
    float g = 0.5f * a * (1.0f + erff(a * 0.70710678118654752f));
    out[i] = g * b;
}

__global__ void k_add(float* __restrict__ x, const float* __restrict__ d, long n)
{
    long i = (long)blockIdx.x * blockDim.x + threadIdx.x;
    if (i < n) x[i] += d[i];
}

// pixel-shuffle r=2: out (Co, Ho, Wo) from in (4*Co, Ho/2, Wo/2)
__global__ void k_pixel_shuffle(const float* __restrict__ in, float* __restrict__ out,
                                int Co, int Ho, int Wo)
{
    long i = (long)blockIdx.x * blockDim.x + threadIdx.x;
    long total = (long)Co * Ho * Wo;
    if (i >= total) return;
    int x  = (int)(i % Wo);
    int y  = (int)((i / Wo) % Ho);
    int co = (int)(i / ((long)Wo * Ho));
    int Hi = Ho >> 1, Wi = Wo >> 1;
    int ii = y & 1, jj = x & 1, h = y >> 1, w = x >> 1;
    int ci = co * 4 + ii * 2 + jj;
    out[i] = in[((long)ci * Hi + h) * Wi + w];
}

__global__ void k_add_clip(const float* __restrict__ inp, const float* __restrict__ res,
                           float* __restrict__ out, long n)
{
    long i = (long)blockIdx.x * blockDim.x + threadIdx.x;
    if (i < n) out[i] = fminf(fmaxf(inp[i] + res[i], 0.0f), 1.0f);
}

// ===========================================================================
// Host-side orchestration
// ===========================================================================
struct BlockP {
    const float *n1, *qkv_w, *qkv_dw, *proj_w, *log_temp, *n2, *pin_w, *dw_w, *pout_w;
};

// Workspace layout (offsets in floats)
static const long OFF_X    = 0L;
static const long OFF_T1   = OFF_X    + 2097152L;
static const long OFF_BIGA = OFF_T1   + 2097152L;
static const long OFF_BIGB = OFF_BIGA + 4718592L;
static const long OFF_ATTO = OFF_BIGB + 4718592L;
static const long OFF_ATTN = OFF_ATTO + 1048576L;
static const long OFF_ENC1 = OFF_ATTN + 32768L;
static const long OFF_ENC2 = OFF_ENC1 + 786432L;
static const long OFF_ENC3 = OFF_ENC2 + 393216L;
static const long OFF_CAT  = OFF_ENC3 + 196608L;
static const long OFF_UPB  = OFF_CAT  + 1572864L;
static const long OFF_SHUF = OFF_UPB  + 1048576L;
static const long OFF_RES  = OFF_SHUF + 1048576L;
static const long OFF_ABF  = OFF_RES  + 65536L;    // 1M bf16  (A operand, padded)
static const long OFF_BTBF = OFF_ABF  + 524288L;   // 2.25M bf16 (B^T operand)
static const long OFF_QBF  = OFF_BTBF + 1179648L;  // 1M bf16
static const long OFF_KBF  = OFF_QBF  + 524288L;   // 1M bf16

static inline long ceilN(long a, long b) { return (a + b - 1) / b * b; }
static inline dim3 g1(long n, int t) { return dim3((unsigned)((n + t - 1) / t)); }

static inline void gemm(const __bf16* A, const __bf16* Bt, float* C,
                        int M, int N, int Kp, int batch,
                        long sA, long sB, long sC, hipStream_t s)
{
    int nT32 = (N + 31) / 32;
    dim3 grid((nT32 + 7) / 8, (M + 31) / 32, batch);
    k_gemm_bf16<<<grid, 256, 0, s>>>(A, Bt, C, M, N, Kp, sA, sB, sC);
}

// 1x1 conv as WMMA GEMM: out(Co x N) = W(Co x Ci) * X(Ci x N)
static void conv1x1(const float* W, const float* X, float* out,
                    int Co, int Ci, long N, float* ws, hipStream_t s)
{
    __bf16* ABF  = (__bf16*)(ws + OFF_ABF);
    __bf16* BTBF = (__bf16*)(ws + OFF_BTBF);
    const int Kp = (int)ceilN(Ci, 32);
    const int Mp = (int)ceilN(Co, 16);
    k_pad_convert_bf16<<<g1((long)Mp * Kp, 256), 256, 0, s>>>(W, ABF, Co, Ci, Mp, Kp);
    dim3 tb(16, 16);
    k_transpose_convert_bf16<<<dim3((unsigned)(N / 16), Kp / 16), tb, 0, s>>>(X, BTBF, Ci, N, Kp);
    gemm(ABF, BTBF, out, Co, (int)N, Kp, 1, 0, 0, 0, s);
}

static void run_block(float* X, const BlockP& p, int C, int H, int W, int heads,
                      float* ws, hipStream_t s)
{
    const long N = (long)H * W;
    float* T1    = ws + OFF_T1;
    float* BIGA  = ws + OFF_BIGA;
    float* BIGB  = ws + OFF_BIGB;
    float* ATTO  = ws + OFF_ATTO;
    float* ATTN  = ws + OFF_ATTN;
    __bf16* ABF  = (__bf16*)(ws + OFF_ABF);
    __bf16* BTBF = (__bf16*)(ws + OFF_BTBF);
    __bf16* QBF  = (__bf16*)(ws + OFF_QBF);
    __bf16* KBF  = (__bf16*)(ws + OFF_KBF);

    // --- MDTA ---
    k_ln2d<<<g1(N, 256), 256, 0, s>>>(X, p.n1, T1, C, N);
    conv1x1(p.qkv_w, T1, BIGA, 3 * C, C, N, ws, s);                               // qkv
    k_dwconv3x3<<<g1((long)3 * C * N, 256), 256, 0, s>>>(BIGA, p.qkv_dw, BIGB, 3 * C, H, W);
    float* q = BIGB;
    float* k = BIGB + (long)C * N;
    float* v = BIGB + 2L * C * N;
    k_l2norm_row_bf16<<<dim3(C), 256, 0, s>>>(q, QBF, N);
    k_l2norm_row_bf16<<<dim3(C), 256, 0, s>>>(k, KBF, N);
    const int hd = C / heads;                                                      // always 48
    // attn[h] = qn * kn^T : (hd x hd), Kp = N (already mult of 32)
    gemm(QBF, KBF, ATTN, hd, hd, (int)N, heads,
         (long)hd * N, (long)hd * N, (long)hd * hd, s);
    k_softmax_temp<<<g1(heads * hd, 128), 128, 0, s>>>(ATTN, p.log_temp, heads, hd);
    // out[h] = attn[h] * v[h] : (hd x N), K = hd(48) -> Kp = 64
    k_pad_convert_bf16<<<g1((long)heads * hd * 64, 256), 256, 0, s>>>(
        ATTN, ABF, heads * hd, hd, heads * hd, 64);
    k_transpose_heads_bf16<<<dim3((unsigned)(N / 16), 4, heads), dim3(16, 16), 0, s>>>(
        v, BTBF, heads, hd, N, 64);
    gemm(ABF, BTBF, ATTO, hd, (int)N, 64, heads,
         (long)hd * 64, (long)N * 64, (long)hd * N, s);
    conv1x1(p.proj_w, ATTO, T1, C, C, N, ws, s);                                   // proj
    k_add<<<g1((long)C * N, 256), 256, 0, s>>>(X, T1, (long)C * N);

    // --- GDFN ---
    k_ln2d<<<g1(N, 256), 256, 0, s>>>(X, p.n2, T1, C, N);
    const int hid = (int)((double)C * 2.66);
    conv1x1(p.pin_w, T1, BIGA, 2 * hid, C, N, ws, s);                              // pin
    k_dwconv3x3<<<g1((long)2 * hid * N, 256), 256, 0, s>>>(BIGA, p.dw_w, BIGB, 2 * hid, H, W);
    k_gelu_mul<<<g1((long)hid * N, 256), 256, 0, s>>>(BIGB, BIGA, hid, N);
    conv1x1(p.pout_w, BIGA, T1, C, hid, N, ws, s);                                 // pout
    k_add<<<g1((long)C * N, 256), 256, 0, s>>>(X, T1, (long)C * N);
}

extern "C" void kernel_launch(void* const* d_in, const int* in_sizes, int n_in,
                              void* d_out, int out_size, void* d_ws, size_t ws_size,
                              hipStream_t stream)
{
    (void)in_sizes; (void)n_in; (void)out_size; (void)ws_size;
    int idx = 0;
    auto nx = [&]() { return (const float*)d_in[idx++]; };
    auto rd_block = [&]() {
        BlockP b;
        b.n1 = nx(); b.qkv_w = nx(); b.qkv_dw = nx(); b.proj_w = nx();
        b.log_temp = nx(); b.n2 = nx(); b.pin_w = nx(); b.dw_w = nx(); b.pout_w = nx();
        return b;
    };

    const float* lr = nx();
    const float* w_patch = nx();
    BlockP enc1[4]; for (int i = 0; i < 4; ++i) enc1[i] = rd_block();
    const float* w_down1 = nx();
    BlockP enc2[6]; for (int i = 0; i < 6; ++i) enc2[i] = rd_block();
    const float* w_down2 = nx();
    BlockP enc3[6]; for (int i = 0; i < 6; ++i) enc3[i] = rd_block();
    const float* w_down3 = nx();
    BlockP enc4[8]; for (int i = 0; i < 8; ++i) enc4[i] = rd_block();
    const float* w_up43 = nx();
    const float* w_red3 = nx();
    BlockP dec3[6]; for (int i = 0; i < 6; ++i) dec3[i] = rd_block();
    const float* w_up32 = nx();
    const float* w_red2 = nx();
    BlockP dec2[6]; for (int i = 0; i < 6; ++i) dec2[i] = rd_block();
    const float* w_up21 = nx();
    const float* w_red1 = nx();
    BlockP dec1[4]; for (int i = 0; i < 4; ++i) dec1[i] = rd_block();
    BlockP refb[4]; for (int i = 0; i < 4; ++i) refb[i] = rd_block();
    const float* w_out = nx();

    float* ws   = (float*)d_ws;
    float* X    = ws + OFF_X;
    float* ENC1 = ws + OFF_ENC1;
    float* ENC2 = ws + OFF_ENC2;
    float* ENC3 = ws + OFF_ENC3;
    float* CAT  = ws + OFF_CAT;
    float* UPB  = ws + OFF_UPB;
    float* SHUF = ws + OFF_SHUF;
    float* RES  = ws + OFF_RES;

    const int c1 = 48, c2 = 96, c3 = 192, c4 = 384;
    const int H1 = 128, H2 = 64, H3 = 32, H4 = 16;
    const long N1 = (long)H1 * H1, N2 = (long)H2 * H2, N3 = (long)H3 * H3;

    // patch embed: 3x3 pad 1, 3 -> 48
    k_conv2d<<<g1((long)c1 * N1, 256), 256, 0, stream>>>(lr, w_patch, X, 3, c1, H1, H1, 3, 3, 1, 1, H1, H1);

    // encoder level 1
    for (int i = 0; i < 4; ++i) run_block(X, enc1[i], c1, H1, H1, 1, ws, stream);
    hipMemcpyAsync(ENC1, X, (size_t)c1 * N1 * sizeof(float), hipMemcpyDeviceToDevice, stream);
    k_conv2d<<<g1((long)c2 * N2, 256), 256, 0, stream>>>(ENC1, w_down1, X, c1, c2, H1, H1, 4, 4, 2, 1, H2, H2);

    // encoder level 2
    for (int i = 0; i < 6; ++i) run_block(X, enc2[i], c2, H2, H2, 2, ws, stream);
    hipMemcpyAsync(ENC2, X, (size_t)c2 * N2 * sizeof(float), hipMemcpyDeviceToDevice, stream);
    k_conv2d<<<g1((long)c3 * N3, 256), 256, 0, stream>>>(ENC2, w_down2, X, c2, c3, H2, H2, 4, 4, 2, 1, H3, H3);

    // encoder level 3
    for (int i = 0; i < 6; ++i) run_block(X, enc3[i], c3, H3, H3, 4, ws, stream);
    hipMemcpyAsync(ENC3, X, (size_t)c3 * N3 * sizeof(float), hipMemcpyDeviceToDevice, stream);
    k_conv2d<<<g1((long)c4 * H4 * H4, 256), 256, 0, stream>>>(ENC3, w_down3, X, c3, c4, H3, H3, 4, 4, 2, 1, H4, H4);

    // latent (level 4)
    for (int i = 0; i < 8; ++i) run_block(X, enc4[i], c4, H4, H4, 8, ws, stream);

    // up 4->3
    k_conv2d<<<g1((long)c3 * 4 * H4 * H4, 256), 256, 0, stream>>>(X, w_up43, UPB, c4, c3 * 4, H4, H4, 3, 3, 1, 1, H4, H4);
    k_pixel_shuffle<<<g1((long)c3 * N3, 256), 256, 0, stream>>>(UPB, SHUF, c3, H3, H3);
    hipMemcpyAsync(CAT, SHUF, (size_t)c3 * N3 * sizeof(float), hipMemcpyDeviceToDevice, stream);
    hipMemcpyAsync(CAT + (long)c3 * N3, ENC3, (size_t)c3 * N3 * sizeof(float), hipMemcpyDeviceToDevice, stream);
    conv1x1(w_red3, CAT, X, c3, 2 * c3, N3, ws, stream);
    for (int i = 0; i < 6; ++i) run_block(X, dec3[i], c3, H3, H3, 4, ws, stream);

    // up 3->2
    k_conv2d<<<g1((long)c2 * 4 * N3, 256), 256, 0, stream>>>(X, w_up32, UPB, c3, c2 * 4, H3, H3, 3, 3, 1, 1, H3, H3);
    k_pixel_shuffle<<<g1((long)c2 * N2, 256), 256, 0, stream>>>(UPB, SHUF, c2, H2, H2);
    hipMemcpyAsync(CAT, SHUF, (size_t)c2 * N2 * sizeof(float), hipMemcpyDeviceToDevice, stream);
    hipMemcpyAsync(CAT + (long)c2 * N2, ENC2, (size_t)c2 * N2 * sizeof(float), hipMemcpyDeviceToDevice, stream);
    conv1x1(w_red2, CAT, X, c2, 2 * c2, N2, ws, stream);
    for (int i = 0; i < 6; ++i) run_block(X, dec2[i], c2, H2, H2, 2, ws, stream);

    // up 2->1
    k_conv2d<<<g1((long)c1 * 4 * N2, 256), 256, 0, stream>>>(X, w_up21, UPB, c2, c1 * 4, H2, H2, 3, 3, 1, 1, H2, H2);
    k_pixel_shuffle<<<g1((long)c1 * N1, 256), 256, 0, stream>>>(UPB, SHUF, c1, H1, H1);
    hipMemcpyAsync(CAT, SHUF, (size_t)c1 * N1 * sizeof(float), hipMemcpyDeviceToDevice, stream);
    hipMemcpyAsync(CAT + (long)c1 * N1, ENC1, (size_t)c1 * N1 * sizeof(float), hipMemcpyDeviceToDevice, stream);
    conv1x1(w_red1, CAT, X, c1, 2 * c1, N1, ws, stream);
    for (int i = 0; i < 4; ++i) run_block(X, dec1[i], c1, H1, H1, 1, ws, stream);

    // refinement
    for (int i = 0; i < 4; ++i) run_block(X, refb[i], c1, H1, H1, 1, ws, stream);

    // output conv + residual + clip
    k_conv2d<<<g1(3L * N1, 256), 256, 0, stream>>>(X, w_out, RES, c1, 3, H1, H1, 3, 3, 1, 1, H1, H1);
    k_add_clip<<<g1(3L * N1, 256), 256, 0, stream>>>(lr, RES, (float*)d_out, 3L * N1);
}
